// Idct2_60043642798469
// MI455X (gfx1250) — matrix operand order repaired and checked by reading, not compiled, not run on hardware
//
#include <hip/hip_runtime.h>

// ---------------------------------------------------------------------------
// 2D IDCT (4096x4096):  out = E @ (X @ D),  D[k][n] = cos(pi*k*(2n+1)/2N),
// E = D^T.  Two 4096^3 GEMMs on the WMMA pipes, bf16 hi/lo split (3 wmma per
// tile) with f32 accumulation for ~fp32 accuracy.
//
// Key restructure: the WMMA B operand needs column-major B == row-major B^T.
//   pass 1:  T  = X @ D   -> B^T = D^T = E   (generated directly)
//   pass 2: out = E @ T   -> B^T = T^T = Tt  (pass 1 stores its result
//                                             transposed, which is free)
// So every tile of every operand is a row-major 128x32 bf16 tile -> all
// staging is done by the Tensor Data Mover (tensor_load_to_lds) with a 2D D#
// whose pad feature reproduces the 40-element padded LDS row stride.
// Double-buffered LDS: DMA tile k+1 while WMMAs consume tile k.
//
// ws layout (6 * 4096*4096 * 2B = 192 MB): [Ehi][Elo][Xhi][Xlo][Tthi][Ttlo]
// ---------------------------------------------------------------------------

#define MDIM 4096
#define NDIM 4096
#define KDIM 4096

typedef __attribute__((ext_vector_type(16))) __bf16 v16bf;
typedef __attribute__((ext_vector_type(8)))  float  v8f;
typedef __attribute__((ext_vector_type(4)))  unsigned int u32x4;
typedef __attribute__((ext_vector_type(8)))  int i32x8;
typedef __attribute__((ext_vector_type(4)))  int i32x4;

union Frag16 { v16bf v; uint4 q[2]; };

#if defined(__AMDGCN__) && __has_builtin(__builtin_amdgcn_tensor_load_to_lds) && \
    __has_builtin(__builtin_amdgcn_s_wait_tensorcnt)
#define USE_TDM 1
#else
#define USE_TDM 0
#endif

__device__ __forceinline__ unsigned short f2bf_bits(float x) {
  unsigned u = __builtin_bit_cast(unsigned, x);
  u = u + 0x7FFFu + ((u >> 16) & 1u);         // round-to-nearest-even
  return (unsigned short)(u >> 16);
}
__device__ __forceinline__ float bf2f(unsigned short s) {
  unsigned u = ((unsigned)s) << 16;
  return __builtin_bit_cast(float, u);
}

#if USE_TDM
// TDM: load a 128-row x 32-elem (bf16) tile, row stride 4096 elems, into LDS
// at byte offset ldsOff, padding +16B after each 64B row (LDS stride 40 elems).
__device__ __forceinline__ void tdm_load_tile(const unsigned short* gsrc,
                                              unsigned ldsOff) {
  unsigned long long ga = (unsigned long long)(const void*)gsrc;
  u32x4 g0;
  g0.x = 1u;                                   // count=1, no gather
  g0.y = ldsOff;                               // lds_addr (bytes)
  g0.z = (unsigned)ga;                         // global_addr[31:0]
  g0.w = ((unsigned)(ga >> 32) & 0x01FFFFFFu)  // global_addr[56:32]
         | 0x80000000u;                        // type=2 ("image")
  i32x8 g1;
  g1[0] = (int)0x06D10000u;  // data_size=2B, pad_en, pad_interval=16DW, pad=4DW
  g1[1] = (int)0x10000000u;  // tensor_dim0 = 4096  (bits 79:48)
  g1[2] = (int)0x10000000u;  // tensor_dim1 = 4096  (bits 111:80)
  g1[3] = (int)0x00200000u;  // tile_dim0   = 32    (bits 127:112)
  g1[4] = 128;               // tile_dim1   = 128   (bits 143:128)
  g1[5] = 4096;              // tensor_dim0_stride = 4096 (bits 207:160)
  g1[6] = 0;
  g1[7] = 0;
  // 6-arg form (clang-23): groups 2,3 unused for 2D tiles.
  __builtin_amdgcn_tensor_load_to_lds(g0, g1, (i32x4)0, (i32x4)0, (i32x8)0, 0);
}
#endif

// ---------------------------------------------------------------------------
// Basis generation: E[i][j] = cos(pi*j*(2i+1)/(2N))  (= D^T).
// Angle reduced mod 4N (=16384) in exact integer arithmetic before cosf.
// ---------------------------------------------------------------------------
__global__ __launch_bounds__(256) void gen_basis(
    unsigned short* __restrict__ Ehi, unsigned short* __restrict__ Elo) {
  unsigned idx = blockIdx.x * 256u + threadIdx.x;
  unsigned i = idx >> 12, j = idx & 4095u;
  const float scale = 3.14159265358979323846f / 8192.0f;  // pi/(2N)
  unsigned p = (j * (2u * i + 1u)) & 16383u;
  float ev = cosf((float)p * scale);
  unsigned short eh = f2bf_bits(ev);
  Ehi[idx] = eh;
  Elo[idx] = f2bf_bits(ev - bf2f(eh));
}

__global__ __launch_bounds__(256) void split_f32_bf16x2(
    const float* __restrict__ x,
    unsigned short* __restrict__ hi, unsigned short* __restrict__ lo) {
  unsigned idx = blockIdx.x * 256u + threadIdx.x;
  float v = x[idx];
  unsigned short h = f2bf_bits(v);
  hi[idx] = h;
  lo[idx] = f2bf_bits(v - bf2f(h));
}

// ---------------------------------------------------------------------------
// C[MxN] = A[MxK] @ Bt[NxK]^T ; A/Bt as bf16 (hi,lo) row-major; f32 acc:
//   C += Ah*Bh + Ah*Bl + Al*Bh
// Block tile 128x128, K-step 32. 8 wave32 waves (4Mx2N); each wave owns a
// 32x64 patch = 2x4 wmma tiles -> 24 v_wmma per K-step per wave.
// Double-buffered dynamic LDS; each buffer holds 4 tiles of [128][40] bf16:
//   +0 Ah, +10240 Al, +20480 Bh, +30720 Bl; buffer stride 40960 B; total 80 KB.
// MODE 0: store f32 C.  MODE 1: store C^T as split bf16 (hi,lo).
// ---------------------------------------------------------------------------
template <int MODE>
__global__ __launch_bounds__(256) void gemm_bf16x3(
    const unsigned short* __restrict__ Ahi, const unsigned short* __restrict__ Alo,
    const unsigned short* __restrict__ Bthi, const unsigned short* __restrict__ Btlo,
    float* __restrict__ Cf,
    unsigned short* __restrict__ Chi, unsigned short* __restrict__ Clo) {
  extern __shared__ unsigned short smem[];     // single dyn-LDS alloc -> offset 0

  const unsigned tid   = threadIdx.x;
  const unsigned lane  = tid & 31u;
  const unsigned wave  = tid >> 5;
  const unsigned waveM = wave >> 1;            // 0..3
  const unsigned waveN = wave & 1u;            // 0..1
  const unsigned m0 = blockIdx.y * 128u;
  const unsigned n0 = blockIdx.x * 128u;

  v8f acc[2][4] = {};

#if !USE_TDM
  const unsigned sRow = tid >> 2;              // 0..63  (+64 for 2nd segment)
  const unsigned sCol = (tid & 3u) << 3;       // 0,8,16,24
#endif

  // ---- stage one K-tile (4 matrices) into LDS buffer `buf` ----
  auto stage = [&](unsigned kt, unsigned buf) {
#if USE_TDM
    if (wave == 0) {                           // one DMA per tile
      const unsigned b = buf * 40960u;
      tdm_load_tile(Ahi  + (size_t)m0 * KDIM + kt, b);
      tdm_load_tile(Alo  + (size_t)m0 * KDIM + kt, b + 10240u);
      tdm_load_tile(Bthi + (size_t)n0 * KDIM + kt, b + 20480u);
      tdm_load_tile(Btlo + (size_t)n0 * KDIM + kt, b + 30720u);
    }
#else
    unsigned short* s = smem + buf * 20480u;
    const size_t gA0 = (size_t)(m0 + sRow) * KDIM + kt + sCol;
    const size_t gA1 = (size_t)(m0 + sRow + 64u) * KDIM + kt + sCol;
    const size_t gB0 = (size_t)(n0 + sRow) * KDIM + kt + sCol;
    const size_t gB1 = (size_t)(n0 + sRow + 64u) * KDIM + kt + sCol;
    const unsigned l0 = sRow * 40u + sCol, l1 = (sRow + 64u) * 40u + sCol;
    *(uint4*)&s[l0]          = *(const uint4*)(Ahi + gA0);
    *(uint4*)&s[l1]          = *(const uint4*)(Ahi + gA1);
    *(uint4*)&s[l0 + 5120u]  = *(const uint4*)(Alo + gA0);
    *(uint4*)&s[l1 + 5120u]  = *(const uint4*)(Alo + gA1);
    *(uint4*)&s[l0 + 10240u] = *(const uint4*)(Bthi + gB0);
    *(uint4*)&s[l1 + 10240u] = *(const uint4*)(Bthi + gB1);
    *(uint4*)&s[l0 + 15360u] = *(const uint4*)(Btlo + gB0);
    *(uint4*)&s[l1 + 15360u] = *(const uint4*)(Btlo + gB1);
#endif
  };

  stage(0u, 0u);                               // prologue: tile 0 -> buffer 0

  unsigned cur = 0u;
  for (unsigned kt = 0; kt < KDIM; kt += 32u) {
#if USE_TDM
    if (wave == 0) __builtin_amdgcn_s_wait_tensorcnt(0);
#endif
    __syncthreads();                           // buf[cur] ready; buf[cur^1] free
    if (kt + 32u < KDIM) stage(kt + 32u, cur ^ 1u);

    unsigned short* sAh = smem + cur * 20480u;
    unsigned short* sAl = sAh + 5120u;
    unsigned short* sBh = sAh + 10240u;
    unsigned short* sBl = sAh + 15360u;

    // ---- operand fragments (ISA 16-bit layouts) ----
    // A lane L: row = L%16; K chunks {0-7,16-23} (L<16) / {8-15,24-31} (L>=16)
    // B lane L: col = L%16; K chunks {0-7,16-23} (L<16) / {8-15,24-31} (L>=16)
    Frag16 ah[2], al[2], bh[4], bl[4];
    const unsigned r16 = lane & 15u;
    const unsigned ka = (lane >> 4) << 3;      // 0 or 8
    const unsigned kb = (lane >> 4) << 3;      // 0 or 8
#pragma unroll
    for (int tm = 0; tm < 2; ++tm) {
      const unsigned row = waveM * 32u + tm * 16u + r16;
      ah[tm].q[0] = *(const uint4*)&sAh[row * 40u + ka];
      ah[tm].q[1] = *(const uint4*)&sAh[row * 40u + ka + 16u];
      al[tm].q[0] = *(const uint4*)&sAl[row * 40u + ka];
      al[tm].q[1] = *(const uint4*)&sAl[row * 40u + ka + 16u];
    }
#pragma unroll
    for (int tn = 0; tn < 4; ++tn) {
      const unsigned col = waveN * 64u + tn * 16u + r16;
      bh[tn].q[0] = *(const uint4*)&sBh[col * 40u + kb];
      bh[tn].q[1] = *(const uint4*)&sBh[col * 40u + kb + 16u];
      bl[tn].q[0] = *(const uint4*)&sBl[col * 40u + kb];
      bl[tn].q[1] = *(const uint4*)&sBl[col * 40u + kb + 16u];
    }

    // ---- 24 wmma per wave per K-step (hi*hi + hi*lo + lo*hi) ----
#pragma unroll
    for (int tm = 0; tm < 2; ++tm)
#pragma unroll
      for (int tn = 0; tn < 4; ++tn) {
        acc[tm][tn] = __builtin_amdgcn_wmma_f32_16x16x32_bf16(
            false, ah[tm].v, false, bh[tn].v, (short)0, acc[tm][tn], false, false);
        acc[tm][tn] = __builtin_amdgcn_wmma_f32_16x16x32_bf16(
            false, ah[tm].v, false, bl[tn].v, (short)0, acc[tm][tn], false, false);
        acc[tm][tn] = __builtin_amdgcn_wmma_f32_16x16x32_bf16(
            false, al[tm].v, false, bh[tn].v, (short)0, acc[tm][tn], false, false);
      }
    cur ^= 1u;
  }

  // ---- epilogue: C/D layout VGPR e -> M = e + (lane>=16 ? 8 : 0), N = lane%16
  const unsigned cr = lane & 15u;
  const unsigned mh = (lane >> 4) << 3;
#pragma unroll
  for (int tm = 0; tm < 2; ++tm)
#pragma unroll
    for (int tn = 0; tn < 4; ++tn) {
      const unsigned col = n0 + waveN * 64u + tn * 16u + cr;
#pragma unroll
      for (int e = 0; e < 8; ++e) {
        const unsigned row = m0 + waveM * 32u + tm * 16u + mh + (unsigned)e;
        float v = acc[tm][tn][e];
        if (MODE == 1) {                       // transposed split-bf16 store
          unsigned short h = f2bf_bits(v);
          Chi[(size_t)col * MDIM + row] = h;
          Clo[(size_t)col * MDIM + row] = f2bf_bits(v - bf2f(h));
        } else {                               // direct f32 store
          Cf[(size_t)row * NDIM + col] = v;
        }
      }
    }
}

// ---------------------------------------------------------------------------
extern "C" void kernel_launch(void* const* d_in, const int* in_sizes, int n_in,
                              void* d_out, int out_size, void* d_ws, size_t ws_size,
                              hipStream_t stream) {
  (void)in_sizes; (void)n_in; (void)out_size; (void)ws_size;

  const float* x = (const float*)d_in[0];
  float* out = (float*)d_out;

  const size_t MAT = (size_t)MDIM * NDIM;       // elements per matrix
  unsigned short* Ehi  = (unsigned short*)d_ws; // needs 6*MAT*2 = 192 MB of ws
  unsigned short* Elo  = Ehi + MAT;
  unsigned short* Xhi  = Elo + MAT;
  unsigned short* Xlo  = Xhi + MAT;
  unsigned short* Tthi = Xlo + MAT;
  unsigned short* Ttlo = Tthi + MAT;

  const unsigned elemBlocks = (unsigned)(MAT / 256);  // 65536

  gen_basis<<<elemBlocks, 256, 0, stream>>>(Ehi, Elo);
  split_f32_bf16x2<<<elemBlocks, 256, 0, stream>>>(x, Xhi, Xlo);

  dim3 grid(NDIM / 128, MDIM / 128), block(256);
  const size_t ldsBytes = 81920;                // 2 buffers x 4 tiles x 10240 B
  // pass 1: Tt = (X @ D)^T, with B^T = E; store transposed split bf16
  gemm_bf16x3<1><<<grid, block, ldsBytes, stream>>>(Xhi, Xlo, Ehi, Elo,
                                                    nullptr, Tthi, Ttlo);
  // pass 2: out = E @ T, with B^T = Tt; store f32
  gemm_bf16x3<0><<<grid, block, ldsBytes, stream>>>(Ehi, Elo, Tthi, Ttlo,
                                                    out, nullptr, nullptr);
}
